// RoiGridPool_48344151884288
// MI455X (gfx1250) — matrix operand order repaired and checked by reading, not compiled, not run on hardware
//
#include <hip/hip_runtime.h>

// ---- problem constants -----------------------------------------------------
#define N_PROP 128
#define N_GRID 216
#define N_KEY  4096
#define C_FEAT 128
#define C_OUT  64
#define G_TOT  (N_PROP * N_GRID)          // 27648 grid points
#define K1PAD  160                        // 131 -> padded to 5*32
#define KB1    5                          // layer-1 K blocks (160/32)
#define KB2    2                          // layer-2 K blocks (64/32)
#define NB_PN  4                          // 64/16 output tiles per branch layer
#define RED_K  (N_GRID * 2 * C_OUT)       // 27648
#define RED_N  256
#define CHUNK_K 128                       // K per TDM-staged chunk
#define GEMM1_CHUNKS (RED_K / CHUNK_K)    // 216

typedef __bf16 bf16_t;
typedef __attribute__((ext_vector_type(16))) __bf16 v16bf;
typedef __attribute__((ext_vector_type(8)))  __bf16 v8bf;
typedef __attribute__((ext_vector_type(8)))  float  v8f;
typedef __attribute__((ext_vector_type(4)))  unsigned int v4u;
typedef __attribute__((ext_vector_type(8)))  int v8i;
typedef __attribute__((ext_vector_type(4)))  int v4i;

static __device__ __forceinline__ v8f wmma_bf16(v16bf a, v16bf b, v8f c) {
  // D = A(16x32 bf16) * B(32x16 bf16) + C(16x16 f32)
  return __builtin_amdgcn_wmma_f32_16x16x32_bf16(false, a, false, b, (short)0, c,
                                                 false, false);
}

// A-fragment K index for element t of the per-lane v16bf (16-bit A 16x32 layout):
// lanes 0-15 hold K {0..7, 16..23}, lanes 16-31 hold K {8..15, 24..31}.
static __device__ __forceinline__ int afrag_k(int t, int hi) {
  return (t < 8) ? (hi * 8 + t) : (16 + hi * 8 + (t - 8));
}

// ---- Tensor Data Mover: 2D tile (tileRows x tileK elems, 2-byte elems) -----
// Builds a D# per CDNA5 ISA §8.3/§8.4 (group0: count/lds/global/type, group1:
// data_size, tensor dims, tile dims, dim0 stride) and issues TENSOR_LOAD_TO_LDS.
// lds_off is the byte offset inside this kernel's (single) static LDS block.
static __device__ __forceinline__ void tdm_load_2d(unsigned lds_off,
                                                   const void* gaddr,
                                                   unsigned tileK,
                                                   unsigned tileRows,
                                                   unsigned rowStrideElems) {
  unsigned long long ga = (unsigned long long)(size_t)gaddr;
  unsigned tensorDim0 = rowStrideElems;   // full row length (elements)
  unsigned tensorDim1 = tileRows;         // rows available from tile start
  v4u g0;
  g0[0] = 1u;                                              // count=1 (user D#)
  g0[1] = lds_off;                                         // lds_addr (bytes)
  g0[2] = (unsigned)(ga & 0xFFFFFFFFu);                    // global_addr[31:0]
  g0[3] = (unsigned)((ga >> 32) & 0x01FFFFFFu)             // global_addr[56:32]
          | (2u << 30);                                    // type = 2 ("image")
  v8i g1;
  g1[0] = (int)(1u << 16);                                 // data_size=1 (2B)
  g1[1] = (int)((tensorDim0 & 0xFFFFu) << 16);             // tensor_dim0[15:0]
  g1[2] = (int)(((tensorDim0 >> 16) & 0xFFFFu)             // tensor_dim0[31:16]
          | ((tensorDim1 & 0xFFFFu) << 16));               // tensor_dim1[15:0]
  g1[3] = (int)(((tensorDim1 >> 16) & 0xFFFFu)             // tensor_dim1[31:16]
          | ((tileK & 0xFFFFu) << 16));                    // tile_dim0
  g1[4] = (int)(tileRows & 0xFFFFu);                       // tile_dim1 (dim2=0)
  g1[5] = (int)(rowStrideElems);                           // dim0_stride[31:0]
  g1[6] = 0;                                               // dim0_stride[47:32]
  g1[7] = 0;                                               // dim1_stride (2D)
  v4i z4 = {0, 0, 0, 0};                                   // groups 2/3 unused
#if defined(__clang_major__) && __clang_major__ >= 23
  v8i z8 = {0, 0, 0, 0, 0, 0, 0, 0};
  __builtin_amdgcn_tensor_load_to_lds(g0, g1, z4, z4, z8, 0);
#else
  __builtin_amdgcn_tensor_load_to_lds(g0, g1, z4, z4, 0);
#endif
}

// ---- weight packing into WMMA B-fragments (small weights) ------------------
// B[k][n] = W[n][k]. Fragment layout [kb][nb][lane][16]; lane = khalf*16+(n%16),
// element t -> K = kb*32 + (t<8 ? khalf*8 + t : 16 + khalf*8 + t-8).
__global__ __launch_bounds__(256) void pack_w_kernel(const float* __restrict__ W,
                                                     int Cout, int Cin, int Kpad,
                                                     bf16_t* __restrict__ out) {
  int total = (Kpad / 32) * (Cout / 16) * 512;
  int i = blockIdx.x * 256 + threadIdx.x;
  if (i >= total) return;
  int t      = i & 15;
  int lane   = (i >> 4) & 31;
  int fragid = i >> 9;
  int nTiles = Cout / 16;
  int kb = fragid / nTiles;
  int nb = fragid % nTiles;
  int n  = nb * 16 + (lane & 15);
  int k  = kb * 32 + ((t < 8) ? ((lane >> 4) * 8 + t) : (16 + (lane >> 4) * 8 + (t - 8)));
  float v = (k < Cin) ? W[n * Cin + k] : 0.0f;
  out[i] = (bf16_t)v;
}

// ---- plain f32 -> bf16 convert (row-major, for TDM-staged weights) ---------
__global__ __launch_bounds__(256) void cvt_bf16_kernel(const float* __restrict__ in,
                                                       bf16_t* __restrict__ out,
                                                       int n) {
  int i = blockIdx.x * 256 + threadIdx.x;
  if (i < n) out[i] = (bf16_t)in[i];
}

// ---- grid point generation -------------------------------------------------
__global__ __launch_bounds__(128) void grid_kernel(const float* __restrict__ wlh,
                                                   const float* __restrict__ center,
                                                   const float* __restrict__ yaw,
                                                   const float* __restrict__ u,
                                                   float* __restrict__ grid) {
  int g = blockIdx.x * 128 + threadIdx.x;
  if (g >= G_TOT) return;
  int n = g / N_GRID;
  float ux = u[g * 3 + 0] * wlh[n * 3 + 0];
  float uy = u[g * 3 + 1] * wlh[n * 3 + 1];
  float uz = u[g * 3 + 2] * wlh[n * 3 + 2];
  float c = __cosf(yaw[n]), s = __sinf(yaw[n]);
  grid[g * 3 + 0] = c * ux - s * uy + center[n * 3 + 0];
  grid[g * 3 + 1] = s * ux + c * uy + center[n * 3 + 1];
  grid[g * 3 + 2] = uz + center[n * 3 + 2];
}

// ---- ball query: first 16 keypoint indices inside each radius --------------
__global__ __launch_bounds__(256) void ballq_kernel(const float* __restrict__ grid,
                                                    const float* __restrict__ kp,
                                                    int* __restrict__ idx0,
                                                    int* __restrict__ idx1) {
  __shared__ float skp[N_KEY * 3];               // 48 KB of 320 KB WGP LDS
  for (int i = threadIdx.x; i < N_KEY * 3; i += 256) skp[i] = kp[i];
  __syncthreads();
  int g = blockIdx.x * 256 + threadIdx.x;        // 27648 = 108 * 256 exactly
  float gx = grid[g * 3 + 0], gy = grid[g * 3 + 1], gz = grid[g * 3 + 2];
  const float r0sq = 0.8f * 0.8f, r1sq = 1.6f * 1.6f;
  int c0 = 0, c1 = 0, f0 = 0, f1 = 0;
  for (int k = 0; k < N_KEY; ++k) {
    float dx = skp[k * 3 + 0] - gx;
    float dy = skp[k * 3 + 1] - gy;
    float dz = skp[k * 3 + 2] - gz;
    float d2 = dx * dx + dy * dy + dz * dz;
    if (d2 < r1sq) {
      if (c1 < 16) { if (c1 == 0) f1 = k; idx1[g * 16 + c1] = k; ++c1; }
      if (d2 < r0sq && c0 < 16) { if (c0 == 0) f0 = k; idx0[g * 16 + c0] = k; ++c0; }
      if (c0 == 16 && c1 == 16) break;
    }
  }
  int fill0 = (c0 > 0) ? f0 : 0;
  int fill1 = (c1 > 0) ? f1 : 0;
  for (int i = c0; i < 16; ++i) idx0[g * 16 + i] = fill0;
  for (int i = c1; i < 16; ++i) idx1[g * 16 + i] = fill1;
}

// ---- branch PointNet MLP: one wave per grid point --------------------------
__global__ __launch_bounds__(128) void pn_kernel(const float* __restrict__ grid,
                                                 const float* __restrict__ kp,
                                                 const float* __restrict__ feats,
                                                 const int*   __restrict__ idx,
                                                 const bf16_t* __restrict__ w1f,
                                                 const float* __restrict__ g1,
                                                 const float* __restrict__ b1,
                                                 const bf16_t* __restrict__ w2f,
                                                 const float* __restrict__ g2,
                                                 const float* __restrict__ b2,
                                                 bf16_t* __restrict__ Abf,
                                                 int choff) {
  __shared__ bf16_t hlds[4][16 * 64];            // per-wave 16x64 h buffer (8 KB)
  const int wave = threadIdx.x >> 5;
  const int lane = threadIdx.x & 31;
  const int g    = blockIdx.x * 4 + wave;        // 27648 = 6912 * 4 exactly
  const int s    = lane & 15;                    // sample index = WMMA M row
  const int hi   = lane >> 4;                    // K-half select of A layout

  int   key = idx[g * 16 + s];
  float gx = grid[g * 3 + 0], gy = grid[g * 3 + 1], gz = grid[g * 3 + 2];
  float dx = kp[key * 3 + 0] - gx;
  float dy = kp[key * 3 + 1] - gy;
  float dz = kp[key * 3 + 2] - gz;

  v8f acc[NB_PN];
  #pragma unroll
  for (int nb = 0; nb < NB_PN; ++nb)
    #pragma unroll
    for (int r = 0; r < 8; ++r) acc[nb][r] = 0.0f;

  // ---- layer 1: 5 K-blocks x 4 N-tiles = 20 WMMA
  #pragma unroll
  for (int kb = 0; kb < KB1; ++kb) {
    v16bf a;
    #pragma unroll
    for (int t = 0; t < 16; ++t) {
      int kk = kb * 32 + afrag_k(t, hi);
      float v;
      if (kk < 3)               v = (kk == 0) ? dx : (kk == 1) ? dy : dz;
      else if (kk < 3 + C_FEAT) v = feats[(kk - 3) * N_KEY + key];
      else                      v = 0.0f;
      a[t] = (bf16_t)v;
    }
    #pragma unroll
    for (int nb = 0; nb < NB_PN; ++nb) {
      v16bf b = *(const v16bf*)(w1f + (((kb * NB_PN + nb) * 32) + lane) * 16);
      acc[nb] = wmma_bf16(a, b, acc[nb]);
    }
  }

  // ---- epilogue 1 + transpose through LDS (C layout -> A layout)
  bf16_t* hme = hlds[wave];
  #pragma unroll
  for (int nb = 0; nb < NB_PN; ++nb) {
    int   o  = nb * 16 + s;
    float gg = g1[o], bb = b1[o];
    #pragma unroll
    for (int r = 0; r < 8; ++r) {
      int   M = r + hi * 8;
      float h = fmaxf(acc[nb][r] * gg + bb, 0.0f);
      hme[M * 64 + o] = (bf16_t)h;
    }
  }
  __syncthreads();   // all 4 waves always active (exact launch geometry)

  // ---- layer 2: 2 K-blocks x 4 N-tiles = 8 WMMA
  v8f acc2[NB_PN];
  #pragma unroll
  for (int nb = 0; nb < NB_PN; ++nb)
    #pragma unroll
    for (int r = 0; r < 8; ++r) acc2[nb][r] = 0.0f;

  #pragma unroll
  for (int kb = 0; kb < KB2; ++kb) {
    v16bf a;
    #pragma unroll
    for (int t = 0; t < 16; ++t) {
      int kk = kb * 32 + afrag_k(t, hi);
      a[t] = hme[s * 64 + kk];                   // ds_load from transposed h
    }
    #pragma unroll
    for (int nb = 0; nb < NB_PN; ++nb) {
      v16bf b = *(const v16bf*)(w2f + (((kb * NB_PN + nb) * 32) + lane) * 16);
      acc2[nb] = wmma_bf16(a, b, acc2[nb]);
    }
  }

  // ---- epilogue 2: relu + max over samples, write permuted reduction input
  const int iGrid = g % N_GRID;
  const int kProp = g / N_GRID;
  #pragma unroll
  for (int nb = 0; nb < NB_PN; ++nb) {
    int   o  = nb * 16 + s;
    float gg = g2[o], bb = b2[o];
    float mx = 0.0f;                             // relu output is >= 0
    #pragma unroll
    for (int r = 0; r < 8; ++r)
      mx = fmaxf(mx, fmaxf(acc2[nb][r] * gg + bb, 0.0f));
    mx = fmaxf(mx, __shfl_xor(mx, 16, 32));      // combine M halves across lanes
    if (lane < 16) {
      int j    = choff + o;                      // pooled channel (0..127)
      int flat = iGrid * (N_PROP * 128) + j * N_PROP + kProp;
      int aRow = flat / RED_K;
      int bCol = flat - aRow * RED_K;
      Abf[(size_t)aRow * RED_K + bCol] = (bf16_t)mx;
    }
  }
}

// ---- GEMM1 via TDM double-buffering: relu(A(128x27648) @ W^T + b) -> bf16 --
// Block = 4 waves = 16 rows x 64 cols of output. Per 128-K chunk, wave 0 DMAs
// A(16x128) and W(64x128) tiles into LDS with TENSOR_LOAD_TO_LDS, overlapped
// with WMMA on the previous chunk (TENSORcnt in-order per wave).
__global__ __launch_bounds__(128) void gemm1_tdm_kernel(const bf16_t* __restrict__ A,
                                                        const bf16_t* __restrict__ W,
                                                        const float* __restrict__ bias,
                                                        bf16_t* __restrict__ out) {
  __shared__ __align__(16) bf16_t smem[2 * (16 + 64) * CHUNK_K];   // 40 KB
  const int wave = threadIdx.x >> 5;
  const int lane = threadIdx.x & 31;
  const int mb   = blockIdx.x >> 2;              // 8 row blocks
  const int nbg  = blockIdx.x & 3;               // 4 col groups of 64
  const int s    = lane & 15;
  const int hi   = lane >> 4;
  const unsigned bufBytes = (16 + 64) * CHUNK_K * 2;   // 20 KB per buffer

  const bf16_t* Ablk = A + (size_t)(mb * 16) * RED_K;
  const bf16_t* Wblk = W + (size_t)(nbg * 64) * RED_K;

  if (wave == 0) {                               // prologue: stage chunk 0
    tdm_load_2d(0, Ablk, CHUNK_K, 16, RED_K);
    tdm_load_2d(16 * CHUNK_K * 2, Wblk, CHUNK_K, 64, RED_K);
  }

  v8f acc;
  #pragma unroll
  for (int r = 0; r < 8; ++r) acc[r] = 0.0f;

  for (int c = 0; c < GEMM1_CHUNKS; ++c) {
    if (wave == 0) {
      if (c + 1 < GEMM1_CHUNKS) {                // issue next chunk, then wait
        unsigned base = ((c + 1) & 1) * bufBytes;  // for current to land
        tdm_load_2d(base, Ablk + (size_t)(c + 1) * CHUNK_K, CHUNK_K, 16, RED_K);
        tdm_load_2d(base + 16 * CHUNK_K * 2,
                    Wblk + (size_t)(c + 1) * CHUNK_K, CHUNK_K, 64, RED_K);
        __builtin_amdgcn_s_wait_tensorcnt((short)2);
      } else {
        __builtin_amdgcn_s_wait_tensorcnt((short)0);
      }
    }
    __syncthreads();                             // publish chunk c to all waves

    const bf16_t* aLds = smem + (size_t)(c & 1) * (16 + 64) * CHUNK_K;
    const bf16_t* wLds = aLds + 16 * CHUNK_K;
    #pragma unroll
    for (int kb = 0; kb < CHUNK_K / 32; ++kb) {
      union { v16bf v; v8bf h[2]; } au;
      const bf16_t* ap = aLds + s * CHUNK_K + kb * 32 + hi * 16;
      au.h[0] = *(const v8bf*)(ap);              // K {base..base+7}
      au.h[1] = *(const v8bf*)(ap + 8);          // K {base+8..base+15}
      // W staged row-major (n x K): one 32B ds read = full B fragment slice
      v16bf b = *(const v16bf*)(wLds + (wave * 16 + s) * CHUNK_K + kb * 32 + hi * 16);
      acc = wmma_bf16(au.v, b, acc);
    }
    __syncthreads();                             // done before buffer overwrite
  }

  int   o  = nbg * 64 + wave * 16 + s;
  float bb = bias[o];
  #pragma unroll
  for (int r = 0; r < 8; ++r) {
    int row = mb * 16 + r + hi * 8;
    out[(size_t)row * RED_N + o] = (bf16_t)fmaxf(acc[r] + bb, 0.0f);
  }
}

// ---- small WMMA GEMM (fragment-packed B): relu(A @ Bfrag + bias) -----------
__global__ __launch_bounds__(128) void gemm_kernel(const bf16_t* __restrict__ A,
                                                   const bf16_t* __restrict__ Bf,
                                                   const float* __restrict__ bias,
                                                   int Kdim, int Nout,
                                                   float*  __restrict__ out_f32,
                                                   bf16_t* __restrict__ out_bf16) {
  const int waveId = (blockIdx.x * blockDim.x + threadIdx.x) >> 5;
  const int lane   = threadIdx.x & 31;
  const int nTiles = Nout / 16;
  const int mb = waveId / nTiles;
  const int nb = waveId % nTiles;
  const int s  = lane & 15;
  const int hi = lane >> 4;
  const int KB = Kdim / 32;

  const bf16_t* arow = A + (size_t)(mb * 16 + s) * Kdim;
  v8f acc;
  #pragma unroll
  for (int r = 0; r < 8; ++r) acc[r] = 0.0f;

  for (int kb = 0; kb < KB; ++kb) {
    union { v16bf v; v8bf h[2]; } au;
    const bf16_t* p0 = arow + kb * 32 + hi * 8;
    au.h[0] = *(const v8bf*)(p0);
    au.h[1] = *(const v8bf*)(p0 + 16);
    const bf16_t* bp = Bf + (((size_t)(kb * nTiles + nb) * 32) + lane) * 16;
    v16bf b = *(const v16bf*)(bp);
    if (kb + 1 < KB)                             // stream next B fragment
      __builtin_prefetch(bp + (size_t)nTiles * 512, 0, 1);
    acc = wmma_bf16(au.v, b, acc);
  }

  int   o  = nb * 16 + s;
  float bb = bias[o];
  #pragma unroll
  for (int r = 0; r < 8; ++r) {
    int   row = mb * 16 + r + hi * 8;
    float v   = fmaxf(acc[r] + bb, 0.0f);
    if (out_f32)  out_f32[(size_t)row * Nout + o]  = v;
    if (out_bf16) out_bf16[(size_t)row * Nout + o] = (bf16_t)v;
  }
}

// ---- host-side orchestration ----------------------------------------------
extern "C" void kernel_launch(void* const* d_in, const int* in_sizes, int n_in,
                              void* d_out, int out_size, void* d_ws, size_t ws_size,
                              hipStream_t stream) {
  (void)in_sizes; (void)n_in; (void)out_size; (void)ws_size;
  const float* wlh     = (const float*)d_in[0];
  const float* center  = (const float*)d_in[1];
  const float* yaw     = (const float*)d_in[2];
  const float* u       = (const float*)d_in[3];
  const float* kp      = (const float*)d_in[4];
  const float* feats   = (const float*)d_in[5];
  const float* pn0_w1  = (const float*)d_in[6];
  const float* pn0_g1  = (const float*)d_in[7];
  const float* pn0_b1  = (const float*)d_in[8];
  const float* pn0_w2  = (const float*)d_in[9];
  const float* pn0_g2  = (const float*)d_in[10];
  const float* pn0_b2  = (const float*)d_in[11];
  const float* pn1_w1  = (const float*)d_in[12];
  const float* pn1_g1  = (const float*)d_in[13];
  const float* pn1_b1  = (const float*)d_in[14];
  const float* pn1_w2  = (const float*)d_in[15];
  const float* pn1_g2  = (const float*)d_in[16];
  const float* pn1_b2  = (const float*)d_in[17];
  const float* red_w1  = (const float*)d_in[18];
  const float* red_b1  = (const float*)d_in[19];
  const float* red_w2  = (const float*)d_in[20];
  const float* red_b2  = (const float*)d_in[21];

  char*  ws  = (char*)d_ws;
  size_t off = 0;
  auto alloc = [&](size_t bytes) -> char* {
    char* p = ws + off;
    off += (bytes + 255) & ~(size_t)255;
    return p;
  };

  float*  grid  = (float*)alloc((size_t)G_TOT * 3 * sizeof(float));
  int*    idx0  = (int*)  alloc((size_t)G_TOT * 16 * sizeof(int));
  int*    idx1  = (int*)  alloc((size_t)G_TOT * 16 * sizeof(int));
  bf16_t* Abf   = (bf16_t*)alloc((size_t)N_PROP * RED_K * sizeof(bf16_t));
  bf16_t* h1bf  = (bf16_t*)alloc((size_t)N_PROP * RED_N * sizeof(bf16_t));
  bf16_t* Wbf   = (bf16_t*)alloc((size_t)RED_N * RED_K * sizeof(bf16_t));
  bf16_t* fw2r  = (bf16_t*)alloc((size_t)(RED_N / 32) * (RED_N / 16) * 512 * 2);
  bf16_t* f0w1  = (bf16_t*)alloc((size_t)KB1 * NB_PN * 512 * 2);
  bf16_t* f0w2  = (bf16_t*)alloc((size_t)KB2 * NB_PN * 512 * 2);
  bf16_t* f1w1  = (bf16_t*)alloc((size_t)KB1 * NB_PN * 512 * 2);
  bf16_t* f1w2  = (bf16_t*)alloc((size_t)KB2 * NB_PN * 512 * 2);

  // 1) weight preparation
  {
    int nW = RED_N * RED_K;                         // 7,077,888
    cvt_bf16_kernel<<<nW / 256, 256, 0, stream>>>(red_w1, Wbf, nW);
    int t2 = (RED_N / 32) * (RED_N / 16) * 512;     // 65,536
    pack_w_kernel<<<t2 / 256, 256, 0, stream>>>(red_w2, RED_N, RED_N, RED_N, fw2r);
    int t3 = KB1 * NB_PN * 512;                     // 10,240
    pack_w_kernel<<<t3 / 256, 256, 0, stream>>>(pn0_w1, C_OUT, C_FEAT + 3, K1PAD, f0w1);
    pack_w_kernel<<<t3 / 256, 256, 0, stream>>>(pn1_w1, C_OUT, C_FEAT + 3, K1PAD, f1w1);
    int t4 = KB2 * NB_PN * 512;                     // 4,096
    pack_w_kernel<<<t4 / 256, 256, 0, stream>>>(pn0_w2, C_OUT, C_OUT, C_OUT, f0w2);
    pack_w_kernel<<<t4 / 256, 256, 0, stream>>>(pn1_w2, C_OUT, C_OUT, C_OUT, f1w2);
  }

  // 2) grid points
  grid_kernel<<<G_TOT / 128, 128, 0, stream>>>(wlh, center, yaw, u, grid);

  // 3) ball query (both radii, keypoints staged in LDS)
  ballq_kernel<<<G_TOT / 256, 256, 0, stream>>>(grid, kp, idx0, idx1);

  // 4) branch PointNets -> permuted reduction input (one wave per grid point)
  pn_kernel<<<G_TOT / 4, 128, 0, stream>>>(grid, kp, feats, idx0,
                                           f0w1, pn0_g1, pn0_b1,
                                           f0w2, pn0_g2, pn0_b2, Abf, 0);
  pn_kernel<<<G_TOT / 4, 128, 0, stream>>>(grid, kp, feats, idx1,
                                           f1w1, pn1_g1, pn1_b1,
                                           f1w2, pn1_g2, pn1_b2, Abf, C_OUT);

  // 5) reduction layer 1: TDM double-buffered WMMA GEMM
  gemm1_tdm_kernel<<<32, 128, 0, stream>>>(Abf, Wbf, red_b1, h1bf);

  // 6) reduction layer 2: (128 x 256) @ (256 x 256) -> relu -> f32 output
  gemm_kernel<<<(N_PROP / 16) * (RED_N / 16) * 32 / 128, 128, 0, stream>>>(
      h1bf, fw2r, red_b2, RED_N, RED_N, (float*)d_out, nullptr);
}
